// MoE_27685359190356
// MI455X (gfx1250) — matrix operand order repaired and checked by reading, not compiled, not run on hardware
//
#include <hip/hip_runtime.h>
#include <hip/hip_bf16.h>
#include <stdint.h>

// ---------------- problem constants ----------------
#define DIMC 2048
#define HC   1408
#define HSC  2816
#define EC   16
#define TC   4096      // B*S tokens
#define NSLOT 8192     // TC * TOPK

// ---------------- GEMM tile config ------------------
#define TM  128        // block tile M
#define TN  64         // block tile N
#define TKc 64         // K step per LDS round (2 x 32-K WMMA sub-steps)
#define TKP 72         // padded LDS row stride (bf16 elems): 144B, 16B-aligned

typedef __attribute__((ext_vector_type(16))) __bf16 v16bf;
typedef __attribute__((ext_vector_type(8)))  float  v8f;

union BF16x16 { v16bf v; __bf16 e[16]; };
union F32x8   { v8f   v; float  e[8];  };

// ---------------- async global->LDS (CDNA5) ----------------
#if defined(__has_builtin)
#if __has_builtin(__builtin_amdgcn_global_load_async_to_lds_b128)
#define HAVE_ASYNC_LDS 1
#endif
#if __has_builtin(__builtin_amdgcn_s_wait_asynccnt)
#define HAVE_WAIT_ASYNC_BUILTIN 1
#endif
#endif
#ifndef HAVE_ASYNC_LDS
#define HAVE_ASYNC_LDS 0
#endif
#ifndef HAVE_WAIT_ASYNC_BUILTIN
#define HAVE_WAIT_ASYNC_BUILTIN 0
#endif

typedef int vi4 __attribute__((vector_size(16)));
typedef __attribute__((address_space(1))) vi4* g1p_vi4;
typedef __attribute__((address_space(3))) vi4* l3p_vi4;

__device__ inline void async_copy_b128(const void* gsrc, void* ldst) {
#if HAVE_ASYNC_LDS
    __builtin_amdgcn_global_load_async_to_lds_b128(
        (g1p_vi4)(uintptr_t)gsrc,
        (l3p_vi4)(unsigned)(uintptr_t)ldst,
        0, 0);
#else
    *(uint4*)ldst = *(const uint4*)gsrc;
#endif
}

__device__ inline void wait_async_zero() {
#if HAVE_ASYNC_LDS
#if HAVE_WAIT_ASYNC_BUILTIN
    __builtin_amdgcn_s_wait_asynccnt(0);
#else
    asm volatile("s_wait_asynccnt 0x0" ::: "memory");
#endif
#endif
}

__device__ inline v8f vzero() {
    F32x8 t;
#pragma unroll
    for (int i = 0; i < 8; ++i) t.e[i] = 0.0f;
    return t.v;
}

__device__ inline v8f wmma_bf16(v16bf a, v16bf b, v8f c) {
    return __builtin_amdgcn_wmma_f32_16x16x32_bf16(
        /*neg_a=*/false, a, /*neg_b=*/false, b,
        /*c_mod=*/(short)0, c, /*reuse_a=*/false, /*reuse_b=*/false);
}

// A fragment: 16x32 bf16, row m = lane&15, K striping per ISA 7.12.2
__device__ inline v16bf lda_frag(const __bf16* base /* As + row0*TKP + ks */, int lane) {
    BF16x16 r;
    const __bf16* p = base + (lane & 15) * TKP + ((lane >> 4) << 3);
#pragma unroll
    for (int i = 0; i < 8; ++i) { r.e[i] = p[i]; r.e[i + 8] = p[16 + i]; }
    return r.v;
}

// B fragment: 32x16 bf16, col n = lane&15, K striping per ISA 7.12.2 (same as A)
__device__ inline v16bf ldb_frag(const __bf16* base /* Bs + col0*TKP + ks */, int lane) {
    BF16x16 r;
    const __bf16* p = base + (lane & 15) * TKP + ((lane >> 4) << 3);
#pragma unroll
    for (int i = 0; i < 8; ++i) { r.e[i] = p[i]; r.e[i + 8] = p[16 + i]; }
    return r.v;
}

// ---------------- LDS tile fills (256 threads, K-extent 64) ----------------
__device__ inline void cvt_store8(__bf16* d, float4 f0, float4 f1) {
    d[0]=(__bf16)f0.x; d[1]=(__bf16)f0.y; d[2]=(__bf16)f0.z; d[3]=(__bf16)f0.w;
    d[4]=(__bf16)f1.x; d[5]=(__bf16)f1.y; d[6]=(__bf16)f1.z; d[7]=(__bf16)f1.w;
}

// A tile 128x64 from fp32, gathered rows via toks[]
__device__ inline void fillA_f32_gather(__bf16* As, const float* __restrict__ X,
                                        const int* toks, int k0, int tid) {
    int r = tid >> 1;
    int h = (tid & 1) << 5;                 // 0 or 32
    const float* src = X + (size_t)toks[r] * DIMC + k0 + h;
    __bf16* d = As + r * TKP + h;
#pragma unroll
    for (int j = 0; j < 4; ++j) {
        float4 f0 = *(const float4*)(src + 8 * j);
        float4 f1 = *(const float4*)(src + 8 * j + 4);
        cvt_store8(d + 8 * j, f0, f1);
    }
}

// A tile 128x64 from fp32, consecutive rows starting row0
__device__ inline void fillA_f32_rows(__bf16* As, const float* __restrict__ X,
                                      size_t ld, int row0, int k0, int tid) {
    int r = tid >> 1;
    int h = (tid & 1) << 5;
    const float* src = X + (size_t)(row0 + r) * ld + k0 + h;
    __bf16* d = As + r * TKP + h;
#pragma unroll
    for (int j = 0; j < 4; ++j) {
        float4 f0 = *(const float4*)(src + 8 * j);
        float4 f1 = *(const float4*)(src + 8 * j + 4);
        cvt_store8(d + 8 * j, f0, f1);
    }
}

// A tile 128x64 from bf16 source: async global->LDS (64B per thread)
__device__ inline void fillA_bf16_rows_async(__bf16* As, const __bf16* __restrict__ X,
                                             size_t ld, int row0, int k0, int tid) {
    int r = tid >> 1;
    int h = (tid & 1) << 5;
    const __bf16* src = X + (size_t)(row0 + r) * ld + k0 + h;
    __bf16* d = As + r * TKP + h;
#pragma unroll
    for (int j = 0; j < 4; ++j)
        async_copy_b128(src + 8 * j, d + 8 * j);
}

// B tile 64x64 from fp32 weights stored [N,K] row-major
__device__ inline void fillB_f32(__bf16* Bs, const float* __restrict__ W,
                                 size_t ld, int n0, int k0, int tid) {
    int n = tid >> 2;
    int q = (tid & 3) << 4;                 // 0,16,32,48
    const float* src = W + (size_t)(n0 + n) * ld + k0 + q;
    __bf16* d = Bs + n * TKP + q;
#pragma unroll
    for (int j = 0; j < 2; ++j) {
        float4 f0 = *(const float4*)(src + 8 * j);
        float4 f1 = *(const float4*)(src + 8 * j + 4);
        cvt_store8(d + 8 * j, f0, f1);
    }
}

__device__ inline float silu_f(float v) {
    return v * __builtin_amdgcn_rcpf(1.0f + __expf(-v));
}

// ============================================================
// Kernel 0: zero control block
// ============================================================
__global__ void moe_zero_ctrl(int* cnt, int* seg, int* fillc,
                              float* imp, float* wsum, float* wtot) {
    int i = threadIdx.x;
    if (i < EC) { cnt[i] = 0; seg[i] = 0; fillc[i] = 0; imp[i] = 0.f; wsum[i] = 0.f; }
    if (i == 0) *wtot = 0.f;
}

// ============================================================
// Kernel 1: gate — one wave per token
// ============================================================
__global__ __launch_bounds__(256)
void moe_gate_kernel(const float* __restrict__ x, const float* __restrict__ gw,
                     int* __restrict__ te, float* __restrict__ tw,
                     float* imp, float* wsum, float* wtot, int* cnt) {
    int lane = threadIdx.x & 31;
    int wv   = threadIdx.x >> 5;
    int t    = blockIdx.x * 8 + wv;

    float xr[DIMC / 32];
    const float* xp = x + (size_t)t * DIMC;
#pragma unroll
    for (int j = 0; j < DIMC / 32; ++j) xr[j] = xp[lane + 32 * j];

    float sc[EC];
#pragma unroll 1
    for (int e = 0; e < EC; ++e) {
        const float* g = gw + (size_t)e * DIMC;
        float p = 0.f;
#pragma unroll
        for (int j = 0; j < DIMC / 32; ++j) p += xr[j] * g[lane + 32 * j];
#pragma unroll
        for (int off = 16; off > 0; off >>= 1) p += __shfl_down(p, off, 32);
        sc[e] = p;   // valid on lane 0
    }

    if (lane == 0) {
        float mx = sc[0];
#pragma unroll
        for (int e = 1; e < EC; ++e) mx = fmaxf(mx, sc[e]);
        float pr[EC]; float se = 0.f;
#pragma unroll
        for (int e = 0; e < EC; ++e) { pr[e] = __expf(sc[e] - mx); se += pr[e]; }
        float inv = __builtin_amdgcn_rcpf(se);
#pragma unroll
        for (int e = 0; e < EC; ++e) pr[e] *= inv;

        int i0 = 0; float v0 = pr[0];
#pragma unroll
        for (int e = 1; e < EC; ++e) if (pr[e] > v0) { v0 = pr[e]; i0 = e; }
        int i1 = -1; float v1 = -1.0f;
#pragma unroll
        for (int e = 0; e < EC; ++e) if (e != i0 && pr[e] > v1) { v1 = pr[e]; i1 = e; }

        float den = v0 + v1 + 1e-9f;
        float rden = __builtin_amdgcn_rcpf(den);
        float w0 = v0 * rden, w1 = v1 * rden;
        te[2 * t] = i0; te[2 * t + 1] = i1;
        tw[2 * t] = w0; tw[2 * t + 1] = w1;
#pragma unroll
        for (int e = 0; e < EC; ++e) atomicAdd(&imp[e], pr[e]);
        atomicAdd(&wsum[i0], w0); atomicAdd(&wsum[i1], w1);
        atomicAdd(wtot, w0 + w1);
        atomicAdd(&cnt[i0], 1);   atomicAdd(&cnt[i1], 1);
    }
}

// ============================================================
// Kernel 2: 16-way exclusive scan (serial, tiny)
// ============================================================
__global__ void moe_scan_kernel(const int* cnt, int* seg) {
    if (threadIdx.x == 0) {
        int c = 0;
        for (int e = 0; e < EC; ++e) { seg[e] = c; c += cnt[e]; }
    }
}

// ============================================================
// Kernel 3: scatter tokens into per-expert slot segments
// ============================================================
__global__ void moe_scatter_kernel(const int* __restrict__ te, const float* __restrict__ tw,
                                   const int* __restrict__ seg, int* fillc,
                                   int* __restrict__ tok_of_slot, float* __restrict__ w_of_slot) {
    int t = blockIdx.x * 256 + threadIdx.x;
    if (t >= TC) return;
#pragma unroll
    for (int k = 0; k < 2; ++k) {
        int e   = te[2 * t + k];
        int pos = atomicAdd(&fillc[e], 1);
        int slot = seg[e] + pos;
        tok_of_slot[slot] = t;
        w_of_slot[slot]   = tw[2 * t + k];
    }
}

// ============================================================
// Kernel 4: shared-expert up-proj + SwiGLU  -> sh (bf16 [T,HS])
// ============================================================
__global__ __launch_bounds__(256)
void moe_shared_up_kernel(const float* __restrict__ x, const float* __restrict__ sw1,
                          const float* __restrict__ sw3, __bf16* __restrict__ sh) {
    int nt = blockIdx.x, mt = blockIdx.y;
    int tid = threadIdx.x, lane = tid & 31, wv = tid >> 5;
    int wr = wv >> 1, wc = wv & 1;
    int row0 = mt * TM, n0 = nt * TN;

    __shared__ __bf16 As[TM * TKP];
    __shared__ __bf16 B1s[TN * TKP];
    __shared__ __bf16 B3s[TN * TKP];

    v8f acc1[2][2], acc3[2][2];
#pragma unroll
    for (int i = 0; i < 2; ++i)
#pragma unroll
        for (int j = 0; j < 2; ++j) { acc1[i][j] = vzero(); acc3[i][j] = vzero(); }

    for (int k0 = 0; k0 < DIMC; k0 += TKc) {
        __syncthreads();
        fillA_f32_rows(As, x, DIMC, row0, k0, tid);
        fillB_f32(B1s, sw1, DIMC, n0, k0, tid);
        fillB_f32(B3s, sw3, DIMC, n0, k0, tid);
        __syncthreads();
#pragma unroll
        for (int ks = 0; ks < TKc; ks += 32) {
            v16bf a0  = lda_frag(As + (wr * 32) * TKP + ks, lane);
            v16bf a1  = lda_frag(As + (wr * 32 + 16) * TKP + ks, lane);
            v16bf b10 = ldb_frag(B1s + (wc * 32) * TKP + ks, lane);
            v16bf b11 = ldb_frag(B1s + (wc * 32 + 16) * TKP + ks, lane);
            v16bf b30 = ldb_frag(B3s + (wc * 32) * TKP + ks, lane);
            v16bf b31 = ldb_frag(B3s + (wc * 32 + 16) * TKP + ks, lane);
            acc1[0][0] = wmma_bf16(a0, b10, acc1[0][0]);
            acc1[0][1] = wmma_bf16(a0, b11, acc1[0][1]);
            acc1[1][0] = wmma_bf16(a1, b10, acc1[1][0]);
            acc1[1][1] = wmma_bf16(a1, b11, acc1[1][1]);
            acc3[0][0] = wmma_bf16(a0, b30, acc3[0][0]);
            acc3[0][1] = wmma_bf16(a0, b31, acc3[0][1]);
            acc3[1][0] = wmma_bf16(a1, b30, acc3[1][0]);
            acc3[1][1] = wmma_bf16(a1, b31, acc3[1][1]);
        }
    }

    int lh = lane >> 4, ln = lane & 15;
#pragma unroll
    for (int mf = 0; mf < 2; ++mf)
#pragma unroll
        for (int nf = 0; nf < 2; ++nf) {
            F32x8 c1; c1.v = acc1[mf][nf];
            F32x8 c3; c3.v = acc3[mf][nf];
#pragma unroll
            for (int r = 0; r < 8; ++r) {
                int m = wr * 32 + mf * 16 + lh * 8 + r;
                int n = n0 + wc * 32 + nf * 16 + ln;
                float val = silu_f(c1.e[r]) * c3.e[r];
                sh[(size_t)(row0 + m) * HSC + n] = (__bf16)val;
            }
        }
}

// ============================================================
// Kernel 5: expert up-proj (w1 & w3, gathered rows) -> hact
// ============================================================
__global__ __launch_bounds__(256)
void moe_expert_up_kernel(const float* __restrict__ x, const float* __restrict__ w1,
                          const float* __restrict__ w3,
                          const int* __restrict__ cnt, const int* __restrict__ seg,
                          const int* __restrict__ tok_of_slot, const float* __restrict__ w_of_slot,
                          __bf16* __restrict__ hact) {
    int e = blockIdx.z, mt = blockIdx.y, nt = blockIdx.x;
    int cntE = cnt[e];
    if (mt * TM >= cntE) return;
    int segE = seg[e];
    int tid = threadIdx.x, lane = tid & 31, wv = tid >> 5;
    int wr = wv >> 1, wc = wv & 1;
    int n0 = nt * TN;

    __shared__ __bf16 As[TM * TKP];
    __shared__ __bf16 B1s[TN * TKP];
    __shared__ __bf16 B3s[TN * TKP];
    __shared__ int   toks[TM];
    __shared__ float tws[TM];

    if (tid < TM) {
        int m = mt * TM + tid;
        bool valid = m < cntE;
        toks[tid] = valid ? tok_of_slot[segE + m] : 0;
        tws[tid]  = valid ? w_of_slot[segE + m]   : 0.f;
    }

    const float* W1 = w1 + (size_t)e * HC * DIMC;
    const float* W3 = w3 + (size_t)e * HC * DIMC;

    v8f acc1[2][2], acc3[2][2];
#pragma unroll
    for (int i = 0; i < 2; ++i)
#pragma unroll
        for (int j = 0; j < 2; ++j) { acc1[i][j] = vzero(); acc3[i][j] = vzero(); }

    for (int k0 = 0; k0 < DIMC; k0 += TKc) {
        __syncthreads();
        fillA_f32_gather(As, x, toks, k0, tid);
        fillB_f32(B1s, W1, DIMC, n0, k0, tid);
        fillB_f32(B3s, W3, DIMC, n0, k0, tid);
        __syncthreads();
#pragma unroll
        for (int ks = 0; ks < TKc; ks += 32) {
            v16bf a0  = lda_frag(As + (wr * 32) * TKP + ks, lane);
            v16bf a1  = lda_frag(As + (wr * 32 + 16) * TKP + ks, lane);
            v16bf b10 = ldb_frag(B1s + (wc * 32) * TKP + ks, lane);
            v16bf b11 = ldb_frag(B1s + (wc * 32 + 16) * TKP + ks, lane);
            v16bf b30 = ldb_frag(B3s + (wc * 32) * TKP + ks, lane);
            v16bf b31 = ldb_frag(B3s + (wc * 32 + 16) * TKP + ks, lane);
            acc1[0][0] = wmma_bf16(a0, b10, acc1[0][0]);
            acc1[0][1] = wmma_bf16(a0, b11, acc1[0][1]);
            acc1[1][0] = wmma_bf16(a1, b10, acc1[1][0]);
            acc1[1][1] = wmma_bf16(a1, b11, acc1[1][1]);
            acc3[0][0] = wmma_bf16(a0, b30, acc3[0][0]);
            acc3[0][1] = wmma_bf16(a0, b31, acc3[0][1]);
            acc3[1][0] = wmma_bf16(a1, b30, acc3[1][0]);
            acc3[1][1] = wmma_bf16(a1, b31, acc3[1][1]);
        }
    }

    int lh = lane >> 4, ln = lane & 15;
#pragma unroll
    for (int mf = 0; mf < 2; ++mf)
#pragma unroll
        for (int nf = 0; nf < 2; ++nf) {
            F32x8 c1; c1.v = acc1[mf][nf];
            F32x8 c3; c3.v = acc3[mf][nf];
#pragma unroll
            for (int r = 0; r < 8; ++r) {
                int m  = wr * 32 + mf * 16 + lh * 8 + r;
                int gm = mt * TM + m;
                if (gm < cntE) {
                    int n = n0 + wc * 32 + nf * 16 + ln;
                    float val = silu_f(c1.e[r]) * c3.e[r] * tws[m];
                    hact[(size_t)(segE + gm) * HC + n] = (__bf16)val;
                }
            }
        }
}

// ============================================================
// Kernel 6: shared-expert down-proj -> writes y (initializes output)
// A tiles stream through async global->LDS (bf16 source)
// ============================================================
__global__ __launch_bounds__(256)
void moe_shared_down_kernel(const __bf16* __restrict__ sh, const float* __restrict__ sw2,
                            float* __restrict__ y) {
    int nt = blockIdx.x, mt = blockIdx.y;
    int tid = threadIdx.x, lane = tid & 31, wv = tid >> 5;
    int wr = wv >> 1, wc = wv & 1;
    int row0 = mt * TM, n0 = nt * TN;

    __shared__ __bf16 As[TM * TKP];
    __shared__ __bf16 Bs[TN * TKP];

    v8f acc[2][2];
#pragma unroll
    for (int i = 0; i < 2; ++i)
#pragma unroll
        for (int j = 0; j < 2; ++j) acc[i][j] = vzero();

    for (int k0 = 0; k0 < HSC; k0 += TKc) {
        __syncthreads();
        fillA_bf16_rows_async(As, sh, HSC, row0, k0, tid);
        fillB_f32(Bs, sw2, HSC, n0, k0, tid);
        wait_async_zero();
        __syncthreads();
#pragma unroll
        for (int ks = 0; ks < TKc; ks += 32) {
            v16bf a0 = lda_frag(As + (wr * 32) * TKP + ks, lane);
            v16bf a1 = lda_frag(As + (wr * 32 + 16) * TKP + ks, lane);
            v16bf b0 = ldb_frag(Bs + (wc * 32) * TKP + ks, lane);
            v16bf b1 = ldb_frag(Bs + (wc * 32 + 16) * TKP + ks, lane);
            acc[0][0] = wmma_bf16(a0, b0, acc[0][0]);
            acc[0][1] = wmma_bf16(a0, b1, acc[0][1]);
            acc[1][0] = wmma_bf16(a1, b0, acc[1][0]);
            acc[1][1] = wmma_bf16(a1, b1, acc[1][1]);
        }
    }

    int lh = lane >> 4, ln = lane & 15;
#pragma unroll
    for (int mf = 0; mf < 2; ++mf)
#pragma unroll
        for (int nf = 0; nf < 2; ++nf) {
            F32x8 c; c.v = acc[mf][nf];
#pragma unroll
            for (int r = 0; r < 8; ++r) {
                int m = wr * 32 + mf * 16 + lh * 8 + r;
                int n = n0 + wc * 32 + nf * 16 + ln;
                y[(size_t)(row0 + m) * DIMC + n] = c.e[r];
            }
        }
}

// ============================================================
// Kernel 7: expert down-proj, atomic scatter-add into y
// A tiles stream through async global->LDS (bf16 source)
// ============================================================
__global__ __launch_bounds__(256)
void moe_expert_down_kernel(const __bf16* __restrict__ hact, const float* __restrict__ w2,
                            const int* __restrict__ cnt, const int* __restrict__ seg,
                            const int* __restrict__ tok_of_slot, float* __restrict__ y) {
    int e = blockIdx.z, mt = blockIdx.y, nt = blockIdx.x;
    int cntE = cnt[e];
    if (mt * TM >= cntE) return;
    int segE = seg[e];
    int tid = threadIdx.x, lane = tid & 31, wv = tid >> 5;
    int wr = wv >> 1, wc = wv & 1;
    int n0 = nt * TN;

    __shared__ __bf16 As[TM * TKP];
    __shared__ __bf16 Bs[TN * TKP];
    __shared__ int toks[TM];

    if (tid < TM) {
        int m = mt * TM + tid;
        toks[tid] = (m < cntE) ? tok_of_slot[segE + m] : 0;
    }

    const float* W2 = w2 + (size_t)e * DIMC * HC;

    v8f acc[2][2];
#pragma unroll
    for (int i = 0; i < 2; ++i)
#pragma unroll
        for (int j = 0; j < 2; ++j) acc[i][j] = vzero();

    for (int k0 = 0; k0 < HC; k0 += TKc) {
        __syncthreads();
        fillA_bf16_rows_async(As, hact, HC, segE + mt * TM, k0, tid);
        fillB_f32(Bs, W2, HC, n0, k0, tid);
        wait_async_zero();
        __syncthreads();
#pragma unroll
        for (int ks = 0; ks < TKc; ks += 32) {
            v16bf a0 = lda_frag(As + (wr * 32) * TKP + ks, lane);
            v16bf a1 = lda_frag(As + (wr * 32 + 16) * TKP + ks, lane);
            v16bf b0 = ldb_frag(Bs + (wc * 32) * TKP + ks, lane);
            v16bf b1 = ldb_frag(Bs + (wc * 32 + 16) * TKP + ks, lane);
            acc[0][0] = wmma_bf16(a0, b0, acc[0][0]);
            acc[0][1] = wmma_bf16(a0, b1, acc[0][1]);
            acc[1][0] = wmma_bf16(a1, b0, acc[1][0]);
            acc[1][1] = wmma_bf16(a1, b1, acc[1][1]);
        }
    }

    int lh = lane >> 4, ln = lane & 15;
#pragma unroll
    for (int mf = 0; mf < 2; ++mf)
#pragma unroll
        for (int nf = 0; nf < 2; ++nf) {
            F32x8 c; c.v = acc[mf][nf];
#pragma unroll
            for (int r = 0; r < 8; ++r) {
                int m  = wr * 32 + mf * 16 + lh * 8 + r;
                int gm = mt * TM + m;
                if (gm < cntE) {
                    int n = n0 + wc * 32 + nf * 16 + ln;
                    atomicAdd(&y[(size_t)toks[m] * DIMC + n], c.e[r]);
                }
            }
        }
}

// ============================================================
// Kernel 8: aux loss finalize
// ============================================================
__global__ void moe_finalize_kernel(const float* imp, const float* wsum, const float* wtot,
                                    float* out_aux) {
    if (threadIdx.x == 0) {
        float tot = *wtot + 1e-9f;
        float s = 0.f;
        for (int e = 0; e < EC; ++e)
            s += (imp[e] / (float)TC) * (wsum[e] / tot);
        *out_aux = (float)EC * s;
    }
}

// ============================================================
// Host launcher
// ============================================================
extern "C" void kernel_launch(void* const* d_in, const int* in_sizes, int n_in,
                              void* d_out, int out_size, void* d_ws, size_t ws_size,
                              hipStream_t stream) {
    const float* x    = (const float*)d_in[0];
    const float* gw   = (const float*)d_in[1];
    const float* w1   = (const float*)d_in[2];
    const float* w2   = (const float*)d_in[3];
    const float* w3   = (const float*)d_in[4];
    const float* sw1  = (const float*)d_in[5];
    const float* sw2  = (const float*)d_in[6];
    const float* sw3  = (const float*)d_in[7];
    float* y = (float*)d_out;                       // [T, DIM] then aux scalar
    float* out_aux = y + (size_t)TC * DIMC;

    char* ws = (char*)d_ws;
    int*   cnt   = (int*)(ws);
    int*   seg   = cnt + EC;
    int*   fillc = seg + EC;
    float* imp   = (float*)(fillc + EC);
    float* wsum  = imp + EC;
    float* wtot  = wsum + EC;
    int*   te    = (int*)  (ws + 512);
    float* twt   = (float*)(ws + 512 + 32768);
    int*   tok_of_slot = (int*)  (ws + 512 + 65536);
    float* w_of_slot   = (float*)(ws + 512 + 98304);
    __bf16* hact = (__bf16*)(ws + 131584);                       // [NSLOT, H] bf16
    __bf16* sh   = hact + (size_t)NSLOT * HC;                    // [T, HS]  bf16

    hipLaunchKernelGGL(moe_zero_ctrl, dim3(1), dim3(128), 0, stream,
                       cnt, seg, fillc, imp, wsum, wtot);
    hipLaunchKernelGGL(moe_gate_kernel, dim3(TC / 8), dim3(256), 0, stream,
                       x, gw, te, twt, imp, wsum, wtot, cnt);
    hipLaunchKernelGGL(moe_scan_kernel, dim3(1), dim3(1), 0, stream, cnt, seg);
    hipLaunchKernelGGL(moe_scatter_kernel, dim3(TC / 256), dim3(256), 0, stream,
                       te, twt, seg, fillc, tok_of_slot, w_of_slot);
    hipLaunchKernelGGL(moe_shared_up_kernel, dim3(HSC / TN, TC / TM), dim3(256), 0, stream,
                       x, sw1, sw3, sh);
    hipLaunchKernelGGL(moe_expert_up_kernel, dim3(HC / TN, TC / TM, EC), dim3(256), 0, stream,
                       x, w1, w3, cnt, seg, tok_of_slot, w_of_slot, hact);
    hipLaunchKernelGGL(moe_shared_down_kernel, dim3(DIMC / TN, TC / TM), dim3(256), 0, stream,
                       sh, sw2, y);
    hipLaunchKernelGGL(moe_expert_down_kernel, dim3(DIMC / TN, TC / TM, EC), dim3(256), 0, stream,
                       hact, w2, cnt, seg, tok_of_slot, y);
    hipLaunchKernelGGL(moe_finalize_kernel, dim3(1), dim3(1), 0, stream,
                       imp, wsum, wtot, out_aux);
}
